// VectorQuantizer_80676665688826
// MI455X (gfx1250) — matrix operand order repaired and checked by reading, not compiled, not run on hardware
//
#include <hip/hip_runtime.h>

// VQ: B=32768, K=8192, D=256, BETA=0.25
#define VQ_B 32768
#define VQ_K 8192
#define VQ_D 256
#define VQ_BETA 0.25f
#define VQ_CH 32                  // columns (codes) staged per LDS chunk
#define VQ_KSPLIT 2               // K sweep split across blocks
#define VQ_KS (VQ_K / VQ_KSPLIT)  // 4096 codes per split
#define VQ_NC (VQ_KS / VQ_CH)     // 128 chunks per split

typedef __attribute__((ext_vector_type(16))) _Float16 v16h;
typedef __attribute__((ext_vector_type(8)))  _Float16 v8h;
typedef __attribute__((ext_vector_type(8)))  float    v8f;

// ---------------------------------------------------------------------------
// Prep: emb f32 -> f16, plus per-code squared norm (exact, in f32).
// ---------------------------------------------------------------------------
__global__ __launch_bounds__(256) void vq_prep_emb(const float* __restrict__ emb,
                                                   _Float16* __restrict__ embh,
                                                   float* __restrict__ enorm) {
    const int k = blockIdx.x;
    const int t = threadIdx.x;
    const float v = emb[(size_t)k * VQ_D + t];
    embh[(size_t)k * VQ_D + t] = (_Float16)v;
    __shared__ float red[256];
    red[t] = v * v;
    __syncthreads();
    for (int s = 128; s > 0; s >>= 1) {
        if (t < s) red[t] += red[t + s];
        __syncthreads();
    }
    if (t == 0) enorm[k] = red[0];
}

// ---------------------------------------------------------------------------
// Prep: z f32 -> f16 (elementwise).
// ---------------------------------------------------------------------------
__global__ __launch_bounds__(256) void vq_prep_z(const float* __restrict__ z,
                                                 _Float16* __restrict__ zh, int n) {
    const int i = blockIdx.x * blockDim.x + threadIdx.x;
    if (i < n) zh[i] = (_Float16)z[i];
}

// ---------------------------------------------------------------------------
// Fused GEMM + argmin, M=32 rows per wave, K sweep split in 2.
//
// Block = 256 threads = 8 waves = 256 rows; blockIdx encodes (rowBlock, split).
// Each block sweeps 4096 codes in 32-column LDS chunks (16 KB, double
// buffered). Each B fragment read from LDS feeds TWO WMMAs (row-sets 0/1),
// halving LDS operand traffic per WMMA vs M=16. Next chunk's global loads are
// issued before compute; one __syncthreads per chunk.
//
// Per-row (dist,idx) candidates for each split go to bestd/bestidx; the
// gather kernel merges (tie -> split 0 = lower index, matching jnp.argmin).
// ---------------------------------------------------------------------------
__global__ __launch_bounds__(256) void vq_argmin(const _Float16* __restrict__ zh,
                                                 const _Float16* __restrict__ embh,
                                                 const float* __restrict__ enorm,
                                                 float* __restrict__ bestd,
                                                 int* __restrict__ bestidx) {
    __shared__ _Float16 sbuf[2][VQ_CH * VQ_D];   // 2 x 16 KB

    const int t        = threadIdx.x;
    const int lane     = t & 31;
    const int wave     = t >> 5;
    const int split    = blockIdx.x & 1;
    const int rowBase  = (blockIdx.x >> 1) * 256 + wave * 32;
    const int codeBase = split * VQ_KS;
    const int r        = lane & 15;
    const int hsel     = lane >> 4;

    // ---- A fragments: 32 rows (two sets), all 8 k-steps, VGPR-resident ----
    const _Float16* zrow0 = zh + (size_t)(rowBase + r) * VQ_D;
    const _Float16* zrow1 = zh + (size_t)(rowBase + 16 + r) * VQ_D;
    v16h a0[8], a1[8];
#pragma unroll
    for (int ks = 0; ks < 8; ++ks) {
        const int o1 = ks * 32 + (hsel << 3);
        const v8h lo0 = *(const v8h*)(zrow0 + o1);
        const v8h hi0 = *(const v8h*)(zrow0 + o1 + 16);
        const v8h lo1 = *(const v8h*)(zrow1 + o1);
        const v8h hi1 = *(const v8h*)(zrow1 + o1 + 16);
#pragma unroll
        for (int i = 0; i < 8; ++i) {
            a0[ks][i] = lo0[i]; a0[ks][8 + i] = hi0[i];
            a1[ks][i] = lo1[i]; a1[ks][8 + i] = hi1[i];
        }
    }

    // ---- stage chunk 0 ----  (chunk = 32*256 halves; thread moves 4 x 16B)
    v8h stage[4];
#pragma unroll
    for (int i = 0; i < 4; ++i)
        stage[i] = *(const v8h*)(embh + (size_t)codeBase * VQ_D + t * 8 + i * 2048);
#pragma unroll
    for (int i = 0; i < 4; ++i)
        *(v8h*)(&sbuf[0][t * 8 + i * 2048]) = stage[i];
    __syncthreads();

    float best0[8], best1[8];
    int   bidx0[8], bidx1[8];
#pragma unroll
    for (int v = 0; v < 8; ++v) {
        best0[v] = 3.4e38f; bidx0[v] = 0;
        best1[v] = 3.4e38f; bidx1[v] = 0;
    }

    const int bofs = hsel << 3;   // B fragment K offset within a k-step

    for (int c = 0; c < VQ_NC; ++c) {
        const int nb = codeBase + c * VQ_CH;

        // -- issue next chunk's global loads (hidden behind the WMMAs) --
        if (c + 1 < VQ_NC) {
            const _Float16* src = embh + (size_t)(nb + VQ_CH) * VQ_D;
#pragma unroll
            for (int i = 0; i < 4; ++i)
                stage[i] = *(const v8h*)(src + t * 8 + i * 2048);
        }

        // -- compute: 2 column tiles, each B fragment feeds 2 WMMAs --
        const _Float16* lbase = &sbuf[c & 1][0];
#pragma unroll
        for (int tile = 0; tile < 2; ++tile) {
            const int colLocal = tile * 16 + r;
            const int col      = nb + colLocal;
            const _Float16* lrow = lbase + colLocal * VQ_D;
            v8f acc0 = {}, acc1 = {};
#pragma unroll
            for (int ks = 0; ks < 8; ++ks) {
                const v8h blo = *(const v8h*)(lrow + ks * 32 + bofs);
                const v8h bhi = *(const v8h*)(lrow + ks * 32 + bofs + 16);
                const v16h b = __builtin_shufflevector(blo, bhi, 0, 1, 2, 3, 4, 5, 6, 7, 8, 9, 10, 11, 12, 13, 14, 15);
                acc0 = __builtin_amdgcn_wmma_f32_16x16x32_f16(
                    false, a0[ks], false, b, (short)0, acc0, false, false);
                acc1 = __builtin_amdgcn_wmma_f32_16x16x32_f16(
                    false, a1[ks], false, b, (short)0, acc1, false, false);
            }
            const float en = enorm[col];
#pragma unroll
            for (int v = 0; v < 8; ++v) {
                const float d0 = en - 2.0f * acc0[v];
                const float d1 = en - 2.0f * acc1[v];
                if (d0 < best0[v]) { best0[v] = d0; bidx0[v] = col; }
                if (d1 < best1[v]) { best1[v] = d1; bidx1[v] = col; }
            }
        }

        // -- commit staged regs to the other buffer, one barrier per chunk --
        if (c + 1 < VQ_NC) {
#pragma unroll
            for (int i = 0; i < 4; ++i)
                *(v8h*)(&sbuf[(c + 1) & 1][t * 8 + i * 2048]) = stage[i];
        }
        __syncthreads();
    }

    // ---- cross-lane argmin within each 16-lane half (masks 1,2,4,8) ----
#pragma unroll
    for (int v = 0; v < 8; ++v) {
        float d = best0[v]; int i = bidx0[v];
        float e = best1[v]; int j = bidx1[v];
#pragma unroll
        for (int m = 1; m < 16; m <<= 1) {
            const float od = __shfl_xor(d, m, 32);
            const int   oi = __shfl_xor(i, m, 32);
            const float oe = __shfl_xor(e, m, 32);
            const int   oj = __shfl_xor(j, m, 32);
            if (od < d || (od == d && oi < i)) { d = od; i = oi; }
            if (oe < e || (oe == e && oj < j)) { e = oe; j = oj; }
        }
        best0[v] = d; bidx0[v] = i;
        best1[v] = e; bidx1[v] = j;
    }
    if (r == 0) {
#pragma unroll
        for (int v = 0; v < 8; ++v) {
            const int row0 = rowBase + hsel * 8 + v;
            const int row1 = row0 + 16;
            bestd[(size_t)split * VQ_B + row0]   = best0[v];
            bestidx[(size_t)split * VQ_B + row0] = bidx0[v];
            bestd[(size_t)split * VQ_B + row1]   = best1[v];
            bestidx[(size_t)split * VQ_B + row1] = bidx1[v];
        }
    }
}

// ---------------------------------------------------------------------------
// Merge splits + gather z_q = emb[idx] (exact f32) + loss partials + indices.
// ---------------------------------------------------------------------------
__global__ __launch_bounds__(256) void vq_gather(const float* __restrict__ z,
                                                 const float* __restrict__ emb,
                                                 const float* __restrict__ bestd,
                                                 const int* __restrict__ bestidx,
                                                 float* __restrict__ zq_out,
                                                 float* __restrict__ idx_out,
                                                 float* __restrict__ accum) {
    const int b = blockIdx.x;
    const int t = threadIdx.x;
    const float d0 = bestd[b];
    const float d1 = bestd[VQ_B + b];
    const int   i0 = bestidx[b];
    const int   i1 = bestidx[VQ_B + b];
    const int  idx = (d1 < d0) ? i1 : i0;   // tie -> split 0 (lower index)

    const float e  = emb[(size_t)idx * VQ_D + t];
    const float zz = z[(size_t)b * VQ_D + t];
    zq_out[(size_t)b * VQ_D + t] = e;       // z_q_st == z_q numerically
    const float diff = e - zz;
    __shared__ float red[256];
    red[t] = diff * diff;
    __syncthreads();
    for (int s = 128; s > 0; s >>= 1) {
        if (t < s) red[t] += red[t + s];
        __syncthreads();
    }
    if (t == 0) {
        atomicAdd(accum, red[0]);
        idx_out[b] = (float)idx;            // <= 8191, exactly representable
    }
}

__global__ void vq_init(float* accum) { *accum = 0.0f; }

__global__ void vq_finalize(const float* __restrict__ accum,
                            float* __restrict__ scalars) {
    const float s = *accum;
    const float v = VQ_BETA * s / (float)((size_t)VQ_B * VQ_D);
    scalars[0] = v;  // vq_loss
    scalars[1] = v;  // commit (same value; stop_gradient only affects grads)
}

// ---------------------------------------------------------------------------
// Launch. Outputs concatenated: z_q_st [B*D] | vq_loss | commit | indices [B].
// Workspace: embh 4MB | zh 16MB | enorm 32KB | bestd 256KB | bestidx 256KB |
//            accum 4B  (~20.7 MB total)
// ---------------------------------------------------------------------------
extern "C" void kernel_launch(void* const* d_in, const int* in_sizes, int n_in,
                              void* d_out, int out_size, void* d_ws, size_t ws_size,
                              hipStream_t stream) {
    const float* z   = (const float*)d_in[0];
    const float* emb = (const float*)d_in[1];

    char* ws = (char*)d_ws;
    _Float16* embh    = (_Float16*)ws;
    _Float16* zh      = (_Float16*)(ws + (size_t)VQ_K * VQ_D * 2);
    float*    enorm   = (float*)(ws + (size_t)VQ_K * VQ_D * 2 + (size_t)VQ_B * VQ_D * 2);
    float*    bestd   = (float*)((char*)enorm + (size_t)VQ_K * 4);
    int*      bestidx = (int*)((char*)bestd + (size_t)VQ_KSPLIT * VQ_B * 4);
    float*    accum   = (float*)((char*)bestidx + (size_t)VQ_KSPLIT * VQ_B * 4);

    float* zq_out  = (float*)d_out;
    float* scalars = zq_out + (size_t)VQ_B * VQ_D;
    float* idx_out = scalars + 2;

    vq_init<<<1, 1, 0, stream>>>(accum);
    vq_prep_emb<<<VQ_K, 256, 0, stream>>>(emb, embh, enorm);
    vq_prep_z<<<(VQ_B * VQ_D) / 256, 256, 0, stream>>>(z, zh, VQ_B * VQ_D);
    vq_argmin<<<(VQ_B / 256) * VQ_KSPLIT, 256, 0, stream>>>(zh, embh, enorm,
                                                            bestd, bestidx);
    vq_gather<<<VQ_B, 256, 0, stream>>>(z, emb, bestd, bestidx,
                                        zq_out, idx_out, accum);
    vq_finalize<<<1, 1, 0, stream>>>(accum, scalars);
}
